// GraphSAGE_57028575756304
// MI455X (gfx1250) — matrix operand order repaired and checked by reading, not compiled, not run on hardware
//
#include <hip/hip_runtime.h>
#include <hip/hip_bf16.h>

typedef __attribute__((ext_vector_type(16))) _Float16 v16h;
typedef __attribute__((ext_vector_type(8)))  float    v8f;

#define N_NODES 100000

// ---------------------------------------------------------------------------
// Edge-phase kernels: degree count + mean-aggregation scatter (L2-resident
// f32 atomics; every feature matrix fits in the 192MB L2).
// 16 consecutive lanes handle consecutive float4s of one src row -> 256B
// contiguous gather per half-wave.
// ---------------------------------------------------------------------------
__global__ __launch_bounds__(256)
void count_edges_kernel(const int* __restrict__ dst, float* __restrict__ cnt, int E) {
  int e = blockIdx.x * blockDim.x + threadIdx.x;
  if (e < E) atomicAdd(&cnt[dst[e]], 1.0f);
}

template<int D>
__global__ __launch_bounds__(256)
void aggregate_edges_kernel(const float* __restrict__ h,
                            const int* __restrict__ src,
                            const int* __restrict__ dst,
                            float* __restrict__ agg, int E) {
  constexpr int G = D / 4;              // float4 groups per row
  int tid = blockIdx.x * blockDim.x + threadIdx.x;
  int total = E * G;
  if (tid >= total) return;
  int e = tid / G;
  int g = tid - e * G;
  int s = src[e];
  int d = dst[e];
  const float4 v = reinterpret_cast<const float4*>(h + (size_t)s * D)[g];
  float* a = agg + (size_t)d * D + g * 4;
  atomicAdd(a + 0, v.x);
  atomicAdd(a + 1, v.y);
  atomicAdd(a + 2, v.z);
  atomicAdd(a + 3, v.w);
}

// ---------------------------------------------------------------------------
// Weight pre-packing: convert W (f32, [DIN,DOUT] row-major) into f16 WMMA
// B fragments laid out per (kc, tn, lane, half) so a GEMM wave loads its
// fragment as one contiguous 32B v16h (two global_load_b128) instead of 16
// strided scalar loads.
// B layout (K x N = 32 x 16, 16-bit): lane L (n = L&15, hi = L>>4),
// half h<8 -> K = 8*hi + h; half h>=8 -> K = 16 + 8*hi + (h-8).
// ---------------------------------------------------------------------------
template<int DIN, int DOUT>
__global__ __launch_bounds__(256)
void pack_b_kernel(const float* __restrict__ W, _Float16* __restrict__ out) {
  constexpr int NT = DOUT / 16;
  constexpr int TOTAL = (DIN / 32) * NT * 32 * 16;
  int tid = blockIdx.x * blockDim.x + threadIdx.x;
  if (tid >= TOTAL) return;
  int h    = tid & 15;
  int lane = (tid >> 4) & 31;
  int ft   = tid >> 9;            // kc*NT + tn
  int tn   = ft % NT;
  int kc   = ft / NT;
  int hi   = lane >> 4;
  int K    = kc * 32 + (h < 8 ? 8 * hi + h : 16 + 8 * hi + (h - 8));
  int col  = tn * 16 + (lane & 15);
  out[tid] = (_Float16)W[(size_t)K * DOUT + col];
}

// ---------------------------------------------------------------------------
// A fragment assembly per the CDNA5 16-bit A (16x32) layout (wave32):
//   lane L (m = L&15, hi = L>>4): halves 0..7  -> K = 8*hi    + 0..7
//                                 halves 8..15 -> K = 16+8*hi + 0..7
// C/D v8f: element r -> row = r + 8*hi, col = lane&15
// ---------------------------------------------------------------------------
__device__ __forceinline__ v16h load_a16(const float* __restrict__ row,
                                         int kbase, int hi, float scale) {
  const float4* p0 = reinterpret_cast<const float4*>(row + kbase + 8 * hi);
  const float4* p1 = reinterpret_cast<const float4*>(row + kbase + 16 + 8 * hi);
  float4 q0 = p0[0], q1 = p0[1];
  float4 q2 = p1[0], q3 = p1[1];
  v16h a;
  a[0]  = (_Float16)(q0.x * scale); a[1]  = (_Float16)(q0.y * scale);
  a[2]  = (_Float16)(q0.z * scale); a[3]  = (_Float16)(q0.w * scale);
  a[4]  = (_Float16)(q1.x * scale); a[5]  = (_Float16)(q1.y * scale);
  a[6]  = (_Float16)(q1.z * scale); a[7]  = (_Float16)(q1.w * scale);
  a[8]  = (_Float16)(q2.x * scale); a[9]  = (_Float16)(q2.y * scale);
  a[10] = (_Float16)(q2.z * scale); a[11] = (_Float16)(q2.w * scale);
  a[12] = (_Float16)(q3.x * scale); a[13] = (_Float16)(q3.y * scale);
  a[14] = (_Float16)(q3.z * scale); a[15] = (_Float16)(q3.w * scale);
  return a;
}

// out = relu?( (agg/max(cnt,1)) @ Wl + b + h @ Wr )
// One wave per 16x16 output tile; K in chunks of 32; two chained WMMAs/chunk.
// B fragments come pre-packed from pack_b_kernel.
template<int DIN, int DOUT, bool RELU>
__global__ __launch_bounds__(256)
void sage_gemm_kernel(const float* __restrict__ agg, const float* __restrict__ cnt,
                      const float* __restrict__ h,
                      const _Float16* __restrict__ BlFrag,
                      const float* __restrict__ bias,
                      const _Float16* __restrict__ BrFrag,
                      float* __restrict__ out, int nTilesM) {
  constexpr int NT = DOUT / 16;
  int wid = blockIdx.x * (blockDim.x >> 5) + (threadIdx.x >> 5);
  if (wid >= nTilesM * NT) return;          // wave-uniform: EXEC stays all-ones
  int tm = wid / NT;
  int tn = wid - tm * NT;
  int lane = threadIdx.x & 31;
  int m  = lane & 15;
  int hi = lane >> 4;
  int row = tm * 16 + m;
  int n0  = tn * 16;

  float invc = 1.0f / fmaxf(cnt[row], 1.0f);

  v8f acc;
  float bv = bias[n0 + m];
#pragma unroll
  for (int r = 0; r < 8; ++r) acc[r] = bv;

  const float* aggRow = agg + (size_t)row * DIN;
  const float* hRow   = h   + (size_t)row * DIN;
  const v16h* blp = reinterpret_cast<const v16h*>(BlFrag);
  const v16h* brp = reinterpret_cast<const v16h*>(BrFrag);

#pragma unroll
  for (int kc = 0; kc < DIN / 32; ++kc) {
    int kbase = kc * 32;
    v16h am = load_a16(aggRow, kbase, hi, invc);   // mean term
    v16h ah = load_a16(hRow,   kbase, hi, 1.0f);   // self term
    v16h bl = blp[(kc * NT + tn) * 32 + lane];     // one 32B contiguous load
    v16h br = brp[(kc * NT + tn) * 32 + lane];
    acc = __builtin_amdgcn_wmma_f32_16x16x32_f16(false, am, false, bl,
                                                 (short)0, acc, false, false);
    acc = __builtin_amdgcn_wmma_f32_16x16x32_f16(false, ah, false, br,
                                                 (short)0, acc, false, false);
  }

#pragma unroll
  for (int r = 0; r < 8; ++r) {
    int orow = tm * 16 + r + 8 * hi;
    float v = acc[r];
    if (RELU) v = fmaxf(v, 0.0f);
    out[(size_t)orow * DOUT + n0 + m] = v;
  }
}

// Layer 3: d_out = 1 -> plain per-node dot products + sigmoid (WMMA would
// waste 15/16 of the tile).
__global__ __launch_bounds__(256)
void sage_layer3_kernel(const float* __restrict__ agg, const float* __restrict__ cnt,
                        const float* __restrict__ h,   const float* __restrict__ W3l,
                        const float* __restrict__ b3,  const float* __restrict__ W3r,
                        float* __restrict__ out, int N) {
  int i = blockIdx.x * blockDim.x + threadIdx.x;
  if (i >= N) return;
  float invc = 1.0f / fmaxf(cnt[i], 1.0f);
  const float* ar = agg + (size_t)i * 32;
  const float* hr = h   + (size_t)i * 32;
  float s = b3[0];
#pragma unroll
  for (int k = 0; k < 32; ++k)
    s += ar[k] * invc * W3l[k] + hr[k] * W3r[k];
  out[i] = 1.0f / (1.0f + __expf(-s));
}

// ---------------------------------------------------------------------------
extern "C" void kernel_launch(void* const* d_in, const int* in_sizes, int n_in,
                              void* d_out, int out_size, void* d_ws, size_t ws_size,
                              hipStream_t stream) {
  const float* x   = (const float*)d_in[0];
  const int*   ei  = (const int*)  d_in[1];   // [2,E] : src row then dst row
  const float* W1l = (const float*)d_in[2];
  const float* b1  = (const float*)d_in[3];
  const float* W1r = (const float*)d_in[4];
  const float* W2l = (const float*)d_in[5];
  const float* b2  = (const float*)d_in[6];
  const float* W2r = (const float*)d_in[7];
  const float* W3l = (const float*)d_in[8];
  const float* b3  = (const float*)d_in[9];
  const float* W3r = (const float*)d_in[10];
  float* out = (float*)d_out;

  const int N = N_NODES;
  const int E = in_sizes[1] / 2;
  const int* src = ei;
  const int* dst = ei + E;

  // Workspace layout (bytes):
  //   cnt  @ 0      (400KB)
  //   agg  @ 1MB    (25.6MB)
  //   h1   @ 29MB   (25.6MB)
  //   h2   @ 57MB   (12.8MB)
  //   B-fragment buffers @ 70MB (24KB total)
  char* base = (char*)d_ws;
  float* cnt = (float*)(base);
  float* agg = (float*)(base + (size_t)(1u  << 20));
  float* h1  = (float*)(base + (size_t)(29u << 20));
  float* h2  = (float*)(base + (size_t)(57u << 20));
  char*  fb  = base + (size_t)(70u << 20);
  _Float16* f1l = (_Float16*)(fb);             // 2kc*4tn*32*16 = 4096 halves
  _Float16* f1r = (_Float16*)(fb + 8192);
  _Float16* f2l = (_Float16*)(fb + 16384);     // 2kc*2tn*32*16 = 2048 halves
  _Float16* f2r = (_Float16*)(fb + 20480);

  const int nTilesM = N / 16;   // 6250 exactly

  // Pre-pack weights into WMMA B-fragment layout (tiny, runs once per call).
  pack_b_kernel<64, 64><<<16, 256, 0, stream>>>(W1l, f1l);
  pack_b_kernel<64, 64><<<16, 256, 0, stream>>>(W1r, f1r);
  pack_b_kernel<64, 32><<<8,  256, 0, stream>>>(W2l, f2l);
  pack_b_kernel<64, 32><<<8,  256, 0, stream>>>(W2r, f2r);

  // Degree counts (shared by all layers)
  hipMemsetAsync(cnt, 0, (size_t)N * sizeof(float), stream);
  count_edges_kernel<<<(E + 255) / 256, 256, 0, stream>>>(dst, cnt, E);

  // ---- Layer 1: x[N,64] -> h1[N,64], ReLU ----
  hipMemsetAsync(agg, 0, (size_t)N * 64 * sizeof(float), stream);
  {
    int total = E * 16;
    aggregate_edges_kernel<64><<<(total + 255) / 256, 256, 0, stream>>>(x, src, dst, agg, E);
  }
  {
    int waves = nTilesM * (64 / 16);
    sage_gemm_kernel<64, 64, true><<<(waves + 7) / 8, 256, 0, stream>>>(
        agg, cnt, x, f1l, b1, f1r, h1, nTilesM);
  }

  // ---- Layer 2: h1[N,64] -> h2[N,32], ReLU ----
  hipMemsetAsync(agg, 0, (size_t)N * 64 * sizeof(float), stream);
  {
    int total = E * 16;
    aggregate_edges_kernel<64><<<(total + 255) / 256, 256, 0, stream>>>(h1, src, dst, agg, E);
  }
  {
    int waves = nTilesM * (32 / 16);
    sage_gemm_kernel<64, 32, true><<<(waves + 7) / 8, 256, 0, stream>>>(
        agg, cnt, h1, f2l, b2, f2r, h2, nTilesM);
  }

  // ---- Layer 3: h2[N,32] -> out[N,1], sigmoid ----
  hipMemsetAsync(agg, 0, (size_t)N * 32 * sizeof(float), stream);
  {
    int total = E * 8;
    aggregate_edges_kernel<32><<<(total + 255) / 256, 256, 0, stream>>>(h2, src, dst, agg, E);
  }
  sage_layer3_kernel<<<(N + 255) / 256, 256, 0, stream>>>(agg, cnt, h2, W3l, b3, W3r, out, N);
}
